// MMORE_DX_11622181503325
// MI455X (gfx1250) — compile-verified, matrix operand order checked by reading, not run on hardware
//
#include <hip/hip_runtime.h>

// ---- problem constants (from reference) ----
#define VV   10000
#define LL   6
#define OO   128
#define EE   128
#define ATD  128
#define DPL  500
#define BB   64
#define TT   32
#define NN   40
#define VP   10016          // V padded to multiple of 32 for WMMA K-tiles
#define NROWS 2048          // B*T
#define NSEG 5              // cooccur V-segments (625 tiles / 5 = 125 each)

typedef __attribute__((ext_vector_type(16))) _Float16 v16h;
typedef __attribute__((ext_vector_type(8)))  float    v8f;

__device__ __forceinline__ v8f wmma16(v16h a, v16h b, v8f c) {
  return __builtin_amdgcn_wmma_f32_16x16x32_f16(false, a, false, b, (short)0, c,
                                                false, false);
}

// A-fragment (16x32 f16) for row-major f32 source; lane: m=lane&15, kb=(lane>>4)*8
// elems 0..7 -> K = k0+kb+e ; elems 8..15 -> K = k0+kb+16+(e-8)   (ISA 7.12.2)
__device__ __forceinline__ v16h fragA_f32(const float* __restrict__ row, int k0, int kb) {
  v16h a;
#pragma unroll
  for (int e = 0; e < 8; ++e) a[e] = (_Float16)row[k0 + kb + e];
#pragma unroll
  for (int e = 0; e < 8; ++e) a[8 + e] = (_Float16)row[k0 + kb + 16 + e];
  return a;
}
// Same addressing for f16 sources (also used for B-fragments from K-contiguous
// transposed weights: lane n=lane&15 picks the row pointer, K runs contiguously).
__device__ __forceinline__ v16h fragH(const _Float16* __restrict__ row, int k0, int kb) {
  v16h a;
#pragma unroll
  for (int e = 0; e < 8; ++e) a[e] = row[k0 + kb + e];
#pragma unroll
  for (int e = 0; e < 8; ++e) a[8 + e] = row[k0 + kb + 16 + e];
  return a;
}

// reductions: hred* reduce independently within each 16-lane half (C-layout rows)
__device__ __forceinline__ float hredSum(float v) {
#pragma unroll
  for (int m = 8; m >= 1; m >>= 1) v += __shfl_xor(v, m, 32);
  return v;
}
__device__ __forceinline__ float hredMax(float v) {
#pragma unroll
  for (int m = 8; m >= 1; m >>= 1) v = fmaxf(v, __shfl_xor(v, m, 32));
  return v;
}
__device__ __forceinline__ float wredSum(float v) {
#pragma unroll
  for (int m = 16; m >= 1; m >>= 1) v += __shfl_xor(v, m, 32);
  return v;
}
__device__ __forceinline__ float wredMax(float v) {
#pragma unroll
  for (int m = 16; m >= 1; m >>= 1) v = fmaxf(v, __shfl_xor(v, m, 32));
  return v;
}

// ---------------- init: zero padding rows/cols ----------------
__global__ void k_init(float* __restrict__ onto_all, _Float16* __restrict__ onto_allT) {
  int i = blockIdx.x * blockDim.x + threadIdx.x;
  if (i < OO) onto_all[(size_t)VV * OO + i] = 0.f;            // padding-row (code==V)
  int j = i - OO;
  if (j >= 0 && j < OO * 16) {                                 // K-tail pad of onto_allT
    int rowc = j >> 4, col = j & 15;
    onto_allT[(size_t)rowc * VP + VV + col] = (_Float16)0.f;
  }
}

// ---------------- f16 transposes of co_W and attn_W ----------------
__global__ void k_transpose(const float* __restrict__ co_W, const float* __restrict__ attn_W,
                            _Float16* __restrict__ co_WT, _Float16* __restrict__ attn_WT) {
  int i = blockIdx.x * blockDim.x + threadIdx.x;
  if (i < EE * VV) {                       // co_W (128,V) row-major -> co_WT (V,128)
    int k = i / VV, v = i % VV;            // coalesced read
    co_WT[(size_t)v * EE + k] = (_Float16)co_W[i];
  } else {
    int j = i - EE * VV;
    if (j < 512 * ATD) {                   // attn_W (512,128) -> attn_WT (128,512)
      int k = j >> 7, n = j & 127;
      attn_WT[(size_t)n * 512 + k] = (_Float16)attn_W[j];
    }
  }
}

// ---------------- stage 1: GRAM ancestor attention (one wave per code) --------
__global__ __launch_bounds__(256) void k_onto(
    const int* __restrict__ leaves, const int* __restrict__ ancestors,
    const float* __restrict__ onto_table,
    const float* __restrict__ Wa, const float* __restrict__ ba,
    const float* __restrict__ Wc, const float* __restrict__ bc,
    float* __restrict__ onto_all, _Float16* __restrict__ onto_allT) {
  __shared__ float sEl[8][LL][OO];
  __shared__ float sEa[8][LL][OO];
  const int tid = threadIdx.x, w = tid >> 5, lane = tid & 31;
  const int v = blockIdx.x * 8 + w;                      // 1250*8 == 10000 exactly
  for (int idx = lane; idx < LL * OO; idx += 32) {
    int l = idx >> 7, c = idx & 127;
    sEl[w][l][c] = onto_table[(size_t)leaves[v * LL + l] * OO + c];
    sEa[w][l][c] = onto_table[(size_t)ancestors[v * LL + l] * OO + c];
  }
  __syncthreads();
  float acc[LL][4];
#pragma unroll
  for (int l = 0; l < LL; ++l)
#pragma unroll
    for (int c = 0; c < 4; ++c) acc[l][c] = 0.f;
  for (int k = 0; k < OO; ++k) {          // leaf half of concat
    float wa[4];
#pragma unroll
    for (int c = 0; c < 4; ++c) wa[c] = Wa[(size_t)k * ATD + lane + 32 * c];
#pragma unroll
    for (int l = 0; l < LL; ++l) {
      float x = sEl[w][l][k];
#pragma unroll
      for (int c = 0; c < 4; ++c) acc[l][c] = fmaf(x, wa[c], acc[l][c]);
    }
  }
  for (int k = 0; k < OO; ++k) {          // ancestor half of concat
    float wa[4];
#pragma unroll
    for (int c = 0; c < 4; ++c) wa[c] = Wa[(size_t)(k + OO) * ATD + lane + 32 * c];
#pragma unroll
    for (int l = 0; l < LL; ++l) {
      float x = sEa[w][l][k];
#pragma unroll
      for (int c = 0; c < 4; ++c) acc[l][c] = fmaf(x, wa[c], acc[l][c]);
    }
  }
  float baR[4], WcR[4];
#pragma unroll
  for (int c = 0; c < 4; ++c) { baR[c] = ba[lane + 32 * c]; WcR[c] = Wc[lane + 32 * c]; }
  float score[LL];
#pragma unroll
  for (int l = 0; l < LL; ++l) {
    float p = 0.f;
#pragma unroll
    for (int c = 0; c < 4; ++c) p += tanhf(acc[l][c] + baR[c]) * WcR[c];
    score[l] = wredSum(p) + bc[0];
  }
  float m = score[0];
#pragma unroll
  for (int l = 1; l < LL; ++l) m = fmaxf(m, score[l]);
  float s = 0.f, aw[LL];
#pragma unroll
  for (int l = 0; l < LL; ++l) { aw[l] = __expf(score[l] - m); s += aw[l]; }
  float inv = 1.f / s;
#pragma unroll
  for (int c = 0; c < 4; ++c) {
    int col = lane + 32 * c;
    float o = 0.f;
#pragma unroll
    for (int l = 0; l < LL; ++l) o += aw[l] * sEa[w][l][col];
    o *= inv;
    onto_all[(size_t)v * OO + col] = o;
    onto_allT[(size_t)col * VP + v] = (_Float16)o;       // K-contiguous f16 for WMMA B
  }
}

// ---------------- stage 2: ontoV = dx_onehot @ onto_all (WMMA, HBM-bound) -----
__global__ __launch_bounds__(128) void k_ontoV(
    const float* __restrict__ dx_onehot, const _Float16* __restrict__ onto_allT,
    float* __restrict__ ontoV) {
  const int tid = threadIdx.x, w = tid >> 5, lane = tid & 31;
  const int m16 = lane & 15, kb = (lane >> 4) * 8;
  const int mt = blockIdx.x;                 // 128 M-tiles of 16 rows
  const int mrow = mt * 16 + m16;            // output row = b*T + t
  const int b = mrow >> 5, t = mrow & 31;
  const float* arow = dx_onehot + ((size_t)t * BB + b) * VV;   // input row t*B+b
  const _Float16* br0 = onto_allT + (size_t)((2 * w + 0) * 16 + m16) * VP;
  const _Float16* br1 = onto_allT + (size_t)((2 * w + 1) * 16 + m16) * VP;
  v8f acc0 = {}, acc1 = {};
  for (int k0 = 0; k0 < 9984; k0 += 32) {
    __builtin_prefetch(arow + k0 + 1024, 0, 1);        // global_prefetch of A stream
    v16h a = fragA_f32(arow, k0, kb);
    acc0 = wmma16(a, fragH(br0, k0, kb), acc0);
    acc1 = wmma16(a, fragH(br1, k0, kb), acc1);
  }
  { // K tail 9984..10015: guard A loads; B is zero-padded to VP
    v16h a;
#pragma unroll
    for (int e = 0; e < 8; ++e) {
      int k = 9984 + kb + e;
      a[e] = (k < VV) ? (_Float16)arow[k] : (_Float16)0.f;
    }
#pragma unroll
    for (int e = 0; e < 8; ++e) {
      int k = 9984 + kb + 16 + e;
      a[8 + e] = (k < VV) ? (_Float16)arow[k] : (_Float16)0.f;
    }
    acc0 = wmma16(a, fragH(br0, 9984, kb), acc0);
    acc1 = wmma16(a, fragH(br1, 9984, kb), acc1);
  }
#pragma unroll
  for (int e = 0; e < 8; ++e) {            // C layout: row = e + half*8, col = lane&15
    int row = mt * 16 + e + (lane >> 4) * 8;
    int c0 = (2 * w) * 16 + m16;
    ontoV[(size_t)row * OO + c0] = acc0[e];
    ontoV[(size_t)row * OO + c0 + 16] = acc1[e];
  }
}

// ---------------- stage 3: ehrV = l2norm(sum_n ehr_table[codes]) --------------
__global__ __launch_bounds__(256) void k_ehrV(const int* __restrict__ dxseqs,
                                              const float* __restrict__ ehr_table,
                                              float* __restrict__ ehrV) {
  const int tid = threadIdx.x, w = tid >> 5, lane = tid & 31;
  const int r = blockIdx.x * 8 + w;                        // 256*8 == 2048
  float acc[4] = {0.f, 0.f, 0.f, 0.f};
  const int* codes = dxseqs + (size_t)r * NN;
  for (int n = 0; n < NN; ++n) {
    const float* row = ehr_table + (size_t)codes[n] * EE;  // code==V -> zero row
#pragma unroll
    for (int c = 0; c < 4; ++c) acc[c] += row[lane + 32 * c];
  }
  float ss = 0.f;
#pragma unroll
  for (int c = 0; c < 4; ++c) ss += acc[c] * acc[c];
  ss = wredSum(ss);
  float inv = 1.f / fmaxf(sqrtf(ss), 1e-12f);
#pragma unroll
  for (int c = 0; c < 4; ++c) ehrV[(size_t)r * EE + lane + 32 * c] = acc[c] * inv;
}

// ------- stage 4: cooccur loss, WMMA + online logsumexp (flash-style) ---------
__global__ __launch_bounds__(32) void k_cooccur(
    const float* __restrict__ ehrV, const _Float16* __restrict__ co_WT,
    const float* __restrict__ co_b, const float* __restrict__ dx_onehot,
    float* __restrict__ partials) {
  const int lane = threadIdx.x;
  const int mt = blockIdx.x, seg = blockIdx.y;
  const int m16 = lane & 15, kb = (lane >> 4) * 8;
  const float* arow = ehrV + (size_t)(mt * 16 + m16) * EE;
  v16h a0 = fragA_f32(arow, 0, kb);       // whole K=128 of A held in registers
  v16h a1 = fragA_f32(arow, 32, kb);
  v16h a2 = fragA_f32(arow, 64, kb);
  v16h a3 = fragA_f32(arow, 96, kb);
  size_t ohoff[8];
#pragma unroll
  for (int e = 0; e < 8; ++e) {
    int row = mt * 16 + e + (lane >> 4) * 8;                // GEMM row = b*T+t
    ohoff[e] = ((size_t)(row & 31) * BB + (row >> 5)) * VV; // onehot row = t*B+b
  }
  float mrun[8], srun[8], dot[8], osum[8];
#pragma unroll
  for (int e = 0; e < 8; ++e) { mrun[e] = -INFINITY; srun[e] = 0.f; dot[e] = 0.f; osum[e] = 0.f; }
  const int nt0 = seg * 125, nt1 = nt0 + 125;               // 625 N-tiles / 5 segs
  for (int nt = nt0; nt < nt1; ++nt) {
    const int n0 = nt * 16;
    const _Float16* brow = co_WT + (size_t)(n0 + m16) * EE;
    v8f acc = {};
    acc = wmma16(a0, fragH(brow, 0, kb), acc);
    acc = wmma16(a1, fragH(brow, 32, kb), acc);
    acc = wmma16(a2, fragH(brow, 64, kb), acc);
    acc = wmma16(a3, fragH(brow, 96, kb), acc);
    const float cb = co_b[n0 + m16];
#pragma unroll
    for (int e = 0; e < 8; ++e) {
      float z = acc[e] + cb;                                // logit
      float cm = hredMax(z);                                // per-row chunk max
      float mnew = fmaxf(mrun[e], cm);
      float cs = hredSum(__expf(z - mnew));
      srun[e] = srun[e] * __expf(mrun[e] - mnew) + cs;      // online rescale
      mrun[e] = mnew;
      float oh = dx_onehot[ohoff[e] + n0 + m16];
      dot[e] = fmaf(oh, z, dot[e]);                         // lane-parallel partials
      osum[e] += oh;
    }
  }
#pragma unroll
  for (int e = 0; e < 8; ++e) { dot[e] = hredSum(dot[e]); osum[e] = hredSum(osum[e]); }
  if (m16 == 0) {
#pragma unroll
    for (int e = 0; e < 8; ++e) {
      int row = mt * 16 + e + (lane >> 4) * 8;
      float* p = partials + ((size_t)row * NSEG + seg) * 4;
      p[0] = mrun[e]; p[1] = srun[e]; p[2] = dot[e]; p[3] = osum[e];
    }
  }
}

__global__ void k_merge(const float* __restrict__ partials, float* __restrict__ contrib) {
  int r = blockIdx.x * blockDim.x + threadIdx.x;
  if (r >= NROWS) return;
  float M = -INFINITY;
  for (int s = 0; s < NSEG; ++s) M = fmaxf(M, partials[((size_t)r * NSEG + s) * 4]);
  float S = 0.f, D = 0.f, Os = 0.f;
  for (int s = 0; s < NSEG; ++s) {
    const float* p = partials + ((size_t)r * NSEG + s) * 4;
    S += p[1] * __expf(p[0] - M);
    D += p[2]; Os += p[3];
  }
  contrib[r] = D - Os * (M + __logf(S));   // sum_v oh*(z - lse)
}

// ------- stage 5: fused attention-MLP (WMMA) + softmax + vs2 + dp head --------
__global__ __launch_bounds__(128) void k_mega(
    const int* __restrict__ dxseqs, const float* __restrict__ ontoV,
    const float* __restrict__ ehrV, const float* __restrict__ ehr_table,
    const float* __restrict__ onto_all, const _Float16* __restrict__ attn_WT,
    const float* __restrict__ attn_b, const float* __restrict__ comb_W,
    const float* __restrict__ dp_W, const float* __restrict__ dp_b,
    float* __restrict__ out) {
  __shared__ _Float16 sA[48 * 520];     // 48x512 attnInput tile, padded stride
  __shared__ float sScoresW[4][48];
  __shared__ float sScores[48];
  __shared__ float sAttn[NN];
  __shared__ float sVec[2 * OO];
  __shared__ float sZ[DPL];
  __shared__ float sRed[4];
  __shared__ float sScal[2];
  __shared__ int   sCodes[NN];
  const int r = blockIdx.x;                    // (b,t): r = b*T + t
  const int tid = threadIdx.x, w = tid >> 5, lane = tid & 31;
  if (tid < NN) sCodes[tid] = dxseqs[(size_t)r * NN + tid];
  __syncthreads();
  // build attnInput = [ontoV | ehrV | dxEHR | dxOnto] in f16 (rows 40..47 zero)
  for (int idx = tid; idx < 48 * 512; idx += 128) {
    int row = idx >> 9, col = idx & 511;
    float val = 0.f;
    if (row < NN) {
      if (col < 128)      val = ontoV[(size_t)r * OO + col];
      else if (col < 256) val = ehrV[(size_t)r * EE + (col - 128)];
      else if (col < 384) val = ehr_table[(size_t)sCodes[row] * EE + (col - 256)];
      else                val = onto_all[(size_t)sCodes[row] * OO + (col - 384)];
    }
    sA[row * 520 + col] = (_Float16)val;
  }
  __syncthreads();
  // MLP GEMM: 3 M-tiles x (2 N-tiles per wave) x 16 K-steps
  const int m16 = lane & 15, kb = (lane >> 4) * 8;
  const _Float16* br0 = attn_WT + (size_t)((2 * w + 0) * 16 + m16) * 512;
  const _Float16* br1 = attn_WT + (size_t)((2 * w + 1) * 16 + m16) * 512;
  const int c0 = (2 * w) * 16 + m16, c1 = c0 + 16;
  const float ab0 = attn_b[c0], ab1 = attn_b[c1];
  const float cw0 = comb_W[c0], cw1 = comb_W[c1];
  float sp[3][8];
#pragma unroll
  for (int mt = 0; mt < 3; ++mt) {
    const _Float16* arow = sA + (mt * 16 + m16) * 520;
    v8f a0 = {}, a1 = {};
#pragma unroll 4
    for (int ks = 0; ks < 16; ++ks) {
      int k0 = ks * 32;
      v16h af = fragH(arow, k0, kb);
      a0 = wmma16(af, fragH(br0, k0, kb), a0);
      a1 = wmma16(af, fragH(br1, k0, kb), a1);
    }
#pragma unroll
    for (int e = 0; e < 8; ++e)
      sp[mt][e] = tanhf(a0[e] + ab0) * cw0 + tanhf(a1[e] + ab1) * cw1;
  }
#pragma unroll
  for (int mt = 0; mt < 3; ++mt)
#pragma unroll
    for (int e = 0; e < 8; ++e) {
      float vsum = hredSum(sp[mt][e]);
      if (m16 == 0) sScoresW[w][mt * 16 + e + (lane >> 4) * 8] = vsum;
    }
  __syncthreads();
  if (tid < 48)                                 // deterministic 4-way merge
    sScores[tid] = sScoresW[0][tid] + sScoresW[1][tid] + sScoresW[2][tid] + sScoresW[3][tid];
  __syncthreads();
  if (tid == 0) {                               // softmax over the 40 codes
    float mx = -INFINITY;
    for (int n = 0; n < NN; ++n) mx = fmaxf(mx, sScores[n]);
    float s = 0.f;
    for (int n = 0; n < NN; ++n) { float e = __expf(sScores[n] - mx); sAttn[n] = e; s += e; }
    float inv = 1.f / s;
    for (int n = 0; n < NN; ++n) sAttn[n] *= inv;
  }
  __syncthreads();
  // vs2 = sum_n attn * dxEmb ; then tanh(l2norm(vs2))
  float ssq = 0.f;
  for (int c = tid; c < 256; c += 128) {
    float a = 0.f;
    for (int n = 0; n < NN; ++n) a += sAttn[n] * (float)sA[n * 520 + 256 + c];
    sVec[c] = a;
    ssq += a * a;
  }
  ssq = wredSum(ssq);
  if (lane == 0) sRed[w] = ssq;
  __syncthreads();
  if (tid == 0) sScal[0] = fmaxf(sqrtf(sRed[0] + sRed[1] + sRed[2] + sRed[3]), 1e-12f);
  __syncthreads();
  float inorm = 1.f / sScal[0];
  for (int c = tid; c < 256; c += 128) sVec[c] = tanhf(sVec[c] * inorm);
  __syncthreads();
  // dp head: GEMV (256x500) + softmax over DPL
  for (int j = tid; j < DPL; j += 128) {
    float z = dp_b[j];
    for (int c = 0; c < 256; ++c) z = fmaf(sVec[c], dp_W[(size_t)c * DPL + j], z);
    sZ[j] = z;
  }
  __syncthreads();
  float lm = -INFINITY;
  for (int j = tid; j < DPL; j += 128) lm = fmaxf(lm, sZ[j]);
  lm = wredMax(lm);
  if (lane == 0) sRed[w] = lm;
  __syncthreads();
  if (tid == 0) sScal[0] = fmaxf(fmaxf(sRed[0], sRed[1]), fmaxf(sRed[2], sRed[3]));
  __syncthreads();
  float M = sScal[0], ls = 0.f;
  for (int j = tid; j < DPL; j += 128) ls += __expf(sZ[j] - M);
  ls = wredSum(ls);
  if (lane == 0) sRed[w] = ls;
  __syncthreads();
  if (tid == 0) sScal[1] = sRed[0] + sRed[1] + sRed[2] + sRed[3];
  __syncthreads();
  float invs = 1.f / sScal[1];
  for (int j = tid; j < DPL; j += 128) out[(size_t)r * DPL + j] = __expf(sZ[j] - M) * invs;
}

// ---------------- finalize loss (deterministic block reduce) ------------------
__global__ __launch_bounds__(256) void k_final(const float* __restrict__ contrib,
                                               float* __restrict__ out) {
  __shared__ float red[256];
  int tid = threadIdx.x;
  float s = 0.f;
  for (int r = tid; r < NROWS; r += 256) s += contrib[r];
  red[tid] = s;
  __syncthreads();
  for (int step = 128; step >= 1; step >>= 1) {
    if (tid < step) red[tid] += red[tid + step];
    __syncthreads();
  }
  if (tid == 0) out[(size_t)NROWS * DPL] = (-10.f / 64.f) * red[0];
}

extern "C" void kernel_launch(void* const* d_in, const int* in_sizes, int n_in,
                              void* d_out, int out_size, void* d_ws, size_t ws_size,
                              hipStream_t stream) {
  (void)in_sizes; (void)n_in; (void)out_size; (void)ws_size;
  const int*   dxseqs    = (const int*)  d_in[0];
  const float* dx_onehot = (const float*)d_in[1];
  const int*   leaves    = (const int*)  d_in[2];
  const int*   ancestors = (const int*)  d_in[3];
  const float* onto_table= (const float*)d_in[4];
  const float* onto_Wa   = (const float*)d_in[5];
  const float* onto_ba   = (const float*)d_in[6];
  const float* onto_Wc   = (const float*)d_in[7];
  const float* onto_bc   = (const float*)d_in[8];
  const float* ehr_table = (const float*)d_in[9];
  const float* attn_W    = (const float*)d_in[10];
  const float* attn_b    = (const float*)d_in[11];
  const float* comb_W    = (const float*)d_in[12];
  const float* dp_W      = (const float*)d_in[16];
  const float* dp_b      = (const float*)d_in[17];
  const float* co_W      = (const float*)d_in[14];
  const float* co_b      = (const float*)d_in[15];
  float* out = (float*)d_out;

  char* ws = (char*)d_ws;
  size_t off = 0;
  auto alloc = [&](size_t bytes) -> void* {
    void* p = ws + off;
    off = (off + bytes + 511) & ~(size_t)511;
    return p;
  };
  float*     onto_all  = (float*)    alloc((size_t)(VV + 1) * OO * 4);
  _Float16*  onto_allT = (_Float16*) alloc((size_t)OO * VP * 2);
  float*     ontoV     = (float*)    alloc((size_t)NROWS * OO * 4);
  float*     ehrV      = (float*)    alloc((size_t)NROWS * EE * 4);
  _Float16*  co_WT     = (_Float16*) alloc((size_t)VV * EE * 2);
  _Float16*  attn_WT   = (_Float16*) alloc((size_t)ATD * 512 * 2);
  float*     partials  = (float*)    alloc((size_t)NROWS * NSEG * 4 * 4);
  float*     contrib   = (float*)    alloc((size_t)NROWS * 4);

  k_init<<<9, 256, 0, stream>>>(onto_all, onto_allT);
  k_transpose<<<(EE * VV + 512 * ATD + 255) / 256, 256, 0, stream>>>(
      co_W, attn_W, co_WT, attn_WT);
  k_onto<<<1250, 256, 0, stream>>>(leaves, ancestors, onto_table, onto_Wa, onto_ba,
                                   onto_Wc, onto_bc, onto_all, onto_allT);
  k_ontoV<<<128, 128, 0, stream>>>(dx_onehot, onto_allT, ontoV);
  k_ehrV<<<256, 256, 0, stream>>>(dxseqs, ehr_table, ehrV);
  k_cooccur<<<dim3(128, NSEG), 32, 0, stream>>>(ehrV, co_WT, co_b, dx_onehot, partials);
  k_merge<<<8, 256, 0, stream>>>(partials, contrib);
  k_mega<<<NROWS, 128, 0, stream>>>(dxseqs, ontoV, ehrV, ehr_table, onto_all,
                                    attn_WT, attn_b, comb_W, dp_W, dp_b, out);
  k_final<<<1, 256, 0, stream>>>(contrib, out);
}
